// BlockKoopmanNet_32916629357072
// MI455X (gfx1250) — matrix operand order repaired and checked by reading, not compiled, not run on hardware
//
#include <hip/hip_runtime.h>

// ---------------------------------------------------------------------------
// BlockKoopmanNet forward for MI455X (gfx1250).
// All matmuls as bf16 WMMA (v_wmma_f32_16x16x32_bf16, f32 accumulate).
// A-tiles staged with GLOBAL_LOAD_ASYNC_TO_LDS_B128 (no VGPR round-trip),
// W-tiles register-transposed into LDS; double-buffered, 1 barrier / K-step.
// B=16384, X=64, U=16, Z=32, H=1024, A=256.
// ---------------------------------------------------------------------------

#if defined(__has_builtin)
#if __has_builtin(__builtin_amdgcn_global_load_async_to_lds_b128)
#define USE_ASYNC_LDS 1
#endif
#endif
#ifndef USE_ASYNC_LDS
#define USE_ASYNC_LDS 0
#endif

typedef __attribute__((ext_vector_type(16))) __bf16 v16bf;
typedef __attribute__((ext_vector_type(8)))  __bf16 v8bf;
typedef __attribute__((ext_vector_type(8)))  float  v8f;

#if USE_ASYNC_LDS
// Builtin signature (from hipcc diagnostic): arg1 = int4 in AS1 (global),
// arg2 = LDS dest, args 3/4 = imm offset / cpol.
typedef __attribute__((__vector_size__(16))) int v4i_t;
typedef __attribute__((address_space(1))) v4i_t* gv4i_p;
typedef __attribute__((address_space(3))) v4i_t* lv4i_p;
#endif

#define BATCH 16384

// ---------------- fp32 -> bf16 conversion ----------------
__global__ __launch_bounds__(256) void f32_to_bf16_k(const float* __restrict__ in,
                                                     __bf16* __restrict__ out, int n) {
  int i = blockIdx.x * 256 + threadIdx.x;
  if (i < n) out[i] = (__bf16)in[i];
}

// ---------------- fused WMMA GEMM: C = epilogue(A @ W + bias) ----------------
// A: M x K bf16 row-major (ld=K).  W: K x N bf16 row-major (ld=N).
// 256 threads = 8 waves, wave grid WR x WC; wave tile 32 x (TN*16).
template<int WR, int WC, int TN, bool SILU_BF16>
__global__ __launch_bounds__(256, 1) void gemm_wmma_bf16(
    const __bf16* __restrict__ A, const __bf16* __restrict__ W,
    const float* __restrict__ bias, void* __restrict__ Cout,
    int M, int N, int K)
{
  constexpr int BM  = WR * 32;
  constexpr int BN  = WC * TN * 16;
  constexpr int BK  = 32;
  constexpr int AST = BK + 16;              // 48 halves = 96B row stride (16B aligned)
  constexpr int WST = BK + 16;
  constexpr int ABUF = BM * AST;
  constexpr int WBUF = BN * WST;
  constexpr int AV = (BM * (BK / 8)) / 256;           // 16B chunks per thread
  constexpr int WCHUNKS = BK * (BN / 8);
  constexpr int WV = (WCHUNKS + 255) / 256;
  constexpr int WQ = BN / 8;

  __shared__ __bf16 sA[2 * ABUF];
  __shared__ __bf16 sWt[2 * WBUF];

  const int tid  = threadIdx.x;
  const int lane = tid & 31;
  const int wave = tid >> 5;
  const int wm   = wave % WR;
  const int wn   = wave / WR;
  const int m0   = blockIdx.y * BM;
  const int n0   = blockIdx.x * BN;

  // CDNA5 16-bit WMMA per-lane fragment addressing.
  const int frow = lane & 15;
  const int fk0  = (lane >> 4) * 8;
  const int fcol = lane & 15;
  const int fkb  = (lane >> 4) * 16;

  v8f acc[2][TN];
  #pragma unroll
  for (int i = 0; i < 2; ++i)
    #pragma unroll
    for (int j = 0; j < TN; ++j)
      acc[i][j] = v8f{0.f, 0.f, 0.f, 0.f, 0.f, 0.f, 0.f, 0.f};

  // ---- per-thread staging descriptors (loop-carried pointers, no mul in loop)
  const __bf16* aSrc[AV];
  int aDst[AV];
  #pragma unroll
  for (int i = 0; i < AV; ++i) {
    const int e  = tid + i * 256;
    const int r  = e >> 2;                  // / (BK/8)
    const int kq = e & 3;
    aSrc[i] = A + (size_t)(m0 + r) * K + kq * 8;
    aDst[i] = r * AST + kq * 8;
  }
  const __bf16* wSrc[WV];
  int wRow[WV], wCol[WV];
  bool wAct[WV];
  #pragma unroll
  for (int i = 0; i < WV; ++i) {
    const int e = tid + i * 256;
    wAct[i] = (WCHUNKS % 256 == 0) || (e < WCHUNKS);
    wRow[i] = e / WQ;
    wCol[i] = e % WQ;
    wSrc[i] = W + (size_t)wRow[i] * N + n0 + wCol[i] * 8;
  }

#if !USE_ASYNC_LDS
  uint4 aReg[AV];
#endif
  uint4 wReg[WV];

  // ---- stage A tile into LDS buffer `buf` (async DMA) / into regs (fallback)
  auto stageA = [&](int buf) {
    #pragma unroll
    for (int i = 0; i < AV; ++i) {
#if USE_ASYNC_LDS
      __builtin_amdgcn_global_load_async_to_lds_b128(
          (gv4i_p)aSrc[i],
          (lv4i_p)&sA[buf * ABUF + aDst[i]],
          0, 0);
#else
      (void)buf;
      aReg[i] = *reinterpret_cast<const uint4*>(aSrc[i]);
#endif
      aSrc[i] += BK;
    }
  };
  auto commitA = [&](int buf) {
#if !USE_ASYNC_LDS
    #pragma unroll
    for (int i = 0; i < AV; ++i)
      *reinterpret_cast<uint4*>(&sA[buf * ABUF + aDst[i]]) = aReg[i];
#else
    (void)buf;
#endif
  };
  auto loadW = [&]() {
    #pragma unroll
    for (int i = 0; i < WV; ++i) {
      if (wAct[i]) wReg[i] = *reinterpret_cast<const uint4*>(wSrc[i]);
      wSrc[i] += (size_t)BK * N;
    }
  };
  auto storeW = [&](int buf) {
    #pragma unroll
    for (int i = 0; i < WV; ++i) {
      if (wAct[i]) {
        const __bf16* h = reinterpret_cast<const __bf16*>(&wReg[i]);
        #pragma unroll
        for (int j = 0; j < 8; ++j)
          sWt[buf * WBUF + (wCol[i] * 8 + j) * WST + wRow[i]] = h[j];
      }
    }
  };
  auto waitStage = [&]() {
#if USE_ASYNC_LDS
#if __has_builtin(__builtin_amdgcn_s_wait_asynccnt)
    __builtin_amdgcn_s_wait_asynccnt(0);
#else
    asm volatile("s_wait_asynccnt 0x0" ::: "memory");
#endif
#endif
  };
  // ---- consume one LDS buffer: fragment loads + WMMA ----
  auto compute = [&](int buf) {
    v16bf afr[2];
    #pragma unroll
    for (int tm = 0; tm < 2; ++tm) {
      const __bf16* base = &sA[buf * ABUF + (wm * 32 + tm * 16 + frow) * AST];
      *reinterpret_cast<v8bf*>(&afr[tm])     = *reinterpret_cast<const v8bf*>(base + fk0);
      *(reinterpret_cast<v8bf*>(&afr[tm])+1) = *reinterpret_cast<const v8bf*>(base + 16 + fk0);
    }
    #pragma unroll
    for (int tn = 0; tn < TN; ++tn) {
      v16bf bfr;
      const __bf16* base = &sWt[buf * WBUF + (wn * TN * 16 + tn * 16 + fcol) * WST + fk0];
      *reinterpret_cast<v8bf*>(&bfr)     = *reinterpret_cast<const v8bf*>(base);
      *(reinterpret_cast<v8bf*>(&bfr)+1) = *reinterpret_cast<const v8bf*>(base + 16);
      #pragma unroll
      for (int tm = 0; tm < 2; ++tm)
        acc[tm][tn] = __builtin_amdgcn_wmma_f32_16x16x32_bf16(
            false, afr[tm], false, bfr, (short)0, acc[tm][tn], false, false);
    }
  };

  // ---- software pipeline over K tiles ----
  const int KT = K / BK;
  stageA(0);
  loadW();
  commitA(0);
  storeW(0);
  waitStage();
  __syncthreads();

  int buf = 0;
  #pragma unroll 2
  for (int kt = 0; kt < KT - 1; ++kt) {
    stageA(buf ^ 1);                        // async DMA next A tile into LDS
    loadW();                                // next W tile: global -> VGPR
    compute(buf);                           // current tile from LDS
    commitA(buf ^ 1);                       // (fallback only)
    storeW(buf ^ 1);
    waitStage();                            // s_wait_asynccnt 0
    __syncthreads();
    buf ^= 1;
  }
  compute(buf);

  // ---- epilogue: bias (+ SiLU -> bf16 | raw f32) ----
  #pragma unroll
  for (int tm = 0; tm < 2; ++tm) {
    #pragma unroll
    for (int tn = 0; tn < TN; ++tn) {
      const int col   = n0 + wn * TN * 16 + tn * 16 + (lane & 15);
      const int rbase = m0 + wm * 32 + tm * 16 + ((lane >> 4) * 8);
      const float bv  = bias[col];
      #pragma unroll
      for (int r = 0; r < 8; ++r) {
        float v = acc[tm][tn][r] + bv;
        if constexpr (SILU_BF16) {
          v = v * __frcp_rn(1.f + __expf(-v));
          reinterpret_cast<__bf16*>(Cout)[(size_t)(rbase + r) * N + col] = (__bf16)v;
        } else {
          reinterpret_cast<float*>(Cout)[(size_t)(rbase + r) * N + col] = v;
        }
      }
    }
  }
}

// ---------------- Koopman state update ----------------
__global__ __launch_bounds__(256) void koopman_update_k(
    const float* __restrict__ z, const float* __restrict__ aux,
    const float* __restrict__ Bmat, const float* __restrict__ u,
    __bf16* __restrict__ znext)
{
  const float DT = 0.02f;
  int t = blockIdx.x * 256 + threadIdx.x;
  if (t >= BATCH * 16) return;
  int b = t >> 4, i = t & 15;

  float z0 = z[b * 32 + 2 * i];
  float z1 = z[b * 32 + 2 * i + 1];
  float av = aux[b * 32 + 2 * i];
  float bv = aux[b * 32 + 2 * i + 1];
  float f  = __expf(av * DT);
  float c  = __cosf(bv * DT);
  float s  = __sinf(bv * DT);
  float az0 = f * (c * z0 - s * z1);
  float az1 = f * (s * z0 + c * z1);

  const float* Br = Bmat + (size_t)b * 512 + (size_t)(2 * i) * 16;
  const float* uv = u + (size_t)b * 16;
  float bu0 = 0.f, bu1 = 0.f;
  #pragma unroll
  for (int j = 0; j < 16; ++j) {
    float uu = uv[j];
    bu0 += Br[j] * uu;
    bu1 += Br[16 + j] * uu;
  }
  znext[b * 32 + 2 * i]     = (__bf16)(z0 + DT * (az0 + bu0));
  znext[b * 32 + 2 * i + 1] = (__bf16)(z1 + DT * (az1 + bu1));
}

// ---------------------------------------------------------------------------
// Host launcher
// ---------------------------------------------------------------------------
static inline void conv_bf16(const float* src, __bf16* dst, int n, hipStream_t s) {
  f32_to_bf16_k<<<(n + 255) / 256, 256, 0, s>>>(src, dst, n);
}

extern "C" void kernel_launch(void* const* d_in, const int* in_sizes, int n_in,
                              void* d_out, int out_size, void* d_ws, size_t ws_size,
                              hipStream_t stream) {
  (void)in_sizes; (void)n_in; (void)out_size; (void)ws_size;
  const int B = BATCH;

  const float* x    = (const float*)d_in[0];
  const float* u    = (const float*)d_in[1];
  const float* e_w1 = (const float*)d_in[2];  const float* e_b1 = (const float*)d_in[3];
  const float* e_w2 = (const float*)d_in[4];  const float* e_b2 = (const float*)d_in[5];
  const float* e_w3 = (const float*)d_in[6];  const float* e_b3 = (const float*)d_in[7];
  const float* d_w1 = (const float*)d_in[8];  const float* d_b1 = (const float*)d_in[9];
  const float* d_w2 = (const float*)d_in[10]; const float* d_b2 = (const float*)d_in[11];
  const float* d_w3 = (const float*)d_in[12]; const float* d_b3 = (const float*)d_in[13];
  const float* d_w4 = (const float*)d_in[14]; const float* d_b4 = (const float*)d_in[15];
  const float* a_w1 = (const float*)d_in[16]; const float* a_b1 = (const float*)d_in[17];
  const float* a_w2 = (const float*)d_in[18]; const float* a_b2 = (const float*)d_in[19];
  const float* a_w3 = (const float*)d_in[20]; const float* a_b3 = (const float*)d_in[21];
  const float* b_w1 = (const float*)d_in[22]; const float* b_b1 = (const float*)d_in[23];
  const float* b_w2 = (const float*)d_in[24]; const float* b_b2 = (const float*)d_in[25];
  const float* b_w3 = (const float*)d_in[26]; const float* b_b3 = (const float*)d_in[27];

  // ---- workspace carving (~115 MB total) ----
  char* base = (char*)d_ws;
  size_t cur = 0;
  auto take = [&](size_t bytes) -> char* {
    char* p = base + cur;
    cur += (bytes + 255) & ~(size_t)255;
    return p;
  };
  __bf16* wx     = (__bf16*)take((size_t)B * 64 * 2);
  __bf16* actA   = (__bf16*)take((size_t)B * 1024 * 2);
  __bf16* actB   = (__bf16*)take((size_t)B * 1024 * 2);
  float*  wz     = (float*)take((size_t)B * 32 * 4);
  float*  waux   = (float*)take((size_t)B * 32 * 4);
  float*  wBmat  = (float*)take((size_t)B * 512 * 4);
  __bf16* wznext = (__bf16*)take((size_t)B * 32 * 2);

  __bf16* ew1b = (__bf16*)take((size_t)64 * 1024 * 2);
  __bf16* ew2b = (__bf16*)take((size_t)1024 * 1024 * 2);
  __bf16* ew3b = (__bf16*)take((size_t)1024 * 32 * 2);
  __bf16* dw1b = (__bf16*)take((size_t)32 * 1024 * 2);
  __bf16* dw2b = (__bf16*)take((size_t)1024 * 1024 * 2);
  __bf16* dw3b = (__bf16*)take((size_t)1024 * 1024 * 2);
  __bf16* dw4b = (__bf16*)take((size_t)1024 * 64 * 2);
  __bf16* aw1b = (__bf16*)take((size_t)64 * 256 * 2);
  __bf16* aw2b = (__bf16*)take((size_t)256 * 256 * 2);
  __bf16* aw3b = (__bf16*)take((size_t)256 * 32 * 2);
  __bf16* bw1b = (__bf16*)take((size_t)64 * 256 * 2);
  __bf16* bw2b = (__bf16*)take((size_t)256 * 256 * 2);
  __bf16* bw3b = (__bf16*)take((size_t)256 * 512 * 2);

  // ---- conversions ----
  conv_bf16(x,    wx,   B * 64,      stream);
  conv_bf16(e_w1, ew1b, 64 * 1024,   stream);
  conv_bf16(e_w2, ew2b, 1024 * 1024, stream);
  conv_bf16(e_w3, ew3b, 1024 * 32,   stream);
  conv_bf16(d_w1, dw1b, 32 * 1024,   stream);
  conv_bf16(d_w2, dw2b, 1024 * 1024, stream);
  conv_bf16(d_w3, dw3b, 1024 * 1024, stream);
  conv_bf16(d_w4, dw4b, 1024 * 64,   stream);
  conv_bf16(a_w1, aw1b, 64 * 256,    stream);
  conv_bf16(a_w2, aw2b, 256 * 256,   stream);
  conv_bf16(a_w3, aw3b, 256 * 32,    stream);
  conv_bf16(b_w1, bw1b, 64 * 256,    stream);
  conv_bf16(b_w2, bw2b, 256 * 256,   stream);
  conv_bf16(b_w3, bw3b, 256 * 512,   stream);

  // Configs: big = 128x128 block tile (wave 32x64), f64 = 128x64, nar = 256x32.
  auto gridBig = [&](int N) { return dim3((unsigned)(N / 128), (unsigned)(B / 128), 1); };
  auto grid64  = [&](int N) { return dim3((unsigned)(N / 64),  (unsigned)(B / 128), 1); };
  auto gridNar = [&](int N) { return dim3((unsigned)(N / 32),  (unsigned)(B / 256), 1); };

  // Encoder: x -> h1 -> h2 -> z
  gemm_wmma_bf16<4, 2, 4, true ><<<gridBig(1024), 256, 0, stream>>>(wx,   ew1b, e_b1, actA, B, 1024, 64);
  gemm_wmma_bf16<4, 2, 4, true ><<<gridBig(1024), 256, 0, stream>>>(actA, ew2b, e_b2, actB, B, 1024, 1024);
  gemm_wmma_bf16<8, 1, 2, false><<<gridNar(32),   256, 0, stream>>>(actB, ew3b, e_b3, wz,   B, 32,   1024);

  // Aux net: x -> 256 -> 256 -> 32
  gemm_wmma_bf16<4, 2, 4, true ><<<gridBig(256),  256, 0, stream>>>(wx,   aw1b, a_b1, actA, B, 256,  64);
  gemm_wmma_bf16<4, 2, 4, true ><<<gridBig(256),  256, 0, stream>>>(actA, aw2b, a_b2, actB, B, 256,  256);
  gemm_wmma_bf16<8, 1, 2, false><<<gridNar(32),   256, 0, stream>>>(actB, aw3b, a_b3, waux, B, 32,   256);

  // B net: x -> 256 -> 256 -> 512 (Bmat, f32)
  gemm_wmma_bf16<4, 2, 4, true ><<<gridBig(256),  256, 0, stream>>>(wx,   bw1b, b_b1, actA, B, 256,  64);
  gemm_wmma_bf16<4, 2, 4, true ><<<gridBig(256),  256, 0, stream>>>(actA, bw2b, b_b2, actB, B, 256,  256);
  gemm_wmma_bf16<4, 2, 4, false><<<gridBig(512),  256, 0, stream>>>(actB, bw3b, b_b3, wBmat,B, 512,  256);

  // Koopman update: z_next = z + DT*(Az + Bu)
  koopman_update_k<<<(B * 16 + 255) / 256, 256, 0, stream>>>(wz, waux, wBmat, u, wznext);

  // Decoder: z_next -> 1024 -> 1024 -> 1024 -> 64 (f32 out)
  gemm_wmma_bf16<4, 2, 4, true ><<<gridBig(1024), 256, 0, stream>>>(wznext, dw1b, d_b1, actA, B, 1024, 32);
  gemm_wmma_bf16<4, 2, 4, true ><<<gridBig(1024), 256, 0, stream>>>(actA,   dw2b, d_b2, actB, B, 1024, 1024);
  gemm_wmma_bf16<4, 2, 4, true ><<<gridBig(1024), 256, 0, stream>>>(actB,   dw3b, d_b3, actA, B, 1024, 1024);
  gemm_wmma_bf16<4, 2, 2, false><<<grid64(64),    256, 0, stream>>>(actA,   dw4b, d_b4, (float*)d_out, B, 64, 1024);
}
